// INR_17824114278639
// MI455X (gfx1250) — matrix operand order repaired and checked by reading, not compiled, not run on hardware
//
#include <hip/hip_runtime.h>
#include <hip/hip_bf16.h>

typedef __attribute__((ext_vector_type(16))) _Float16 v16h;
typedef __attribute__((ext_vector_type(8)))  float    v8f;

#define NLV    16              // levels
#define LOG2T  19
#define TSIZE  (1u << LOG2T)   // hash entries per level
#define WIDTH  64
#define OUTD   16

#define WMMA_F16(a, b, c) \
    __builtin_amdgcn_wmma_f32_16x16x32_f16(false, (a), false, (b), (short)0, (c), false, false)

// Build a layer-(i+1) B fragment (K = 32 neurons of chunk, N = 16 points) from two
// layer-i D fragments (dE = neuron tile 2c, dO = neuron tile 2c+1), f32 -> f16.
// D layout: lane = point (lane&15), vgpr r = neuron 16*tile + r + (laneHi?8:0).
// B layout: lane = point, element e  = neuron 32*c + e + (laneHi?16:0).
__device__ __forceinline__ v16h make_bfrag(const float* dE, const float* dO, int hi) {
    v16h b;
#pragma unroll
    for (int e = 0; e < 8; ++e) {
        float sE = __shfl_xor(dE[e], 16, 32);
        float sO = __shfl_xor(dO[e], 16, 32);
        float lo = hi ? sO    : dE[e];   // elements 0..7
        float hv = hi ? dO[e] : sE;      // elements 8..15
        b[e]     = (_Float16)lo;
        b[e + 8] = (_Float16)hv;
    }
    return b;
}

__global__ __launch_bounds__(128) void ngp_fused_kernel(
    const float* __restrict__ xyz,
    const float* __restrict__ bb,
    const float* __restrict__ table,
    const float* __restrict__ W1, const float* __restrict__ B1,
    const float* __restrict__ W2, const float* __restrict__ B2,
    const float* __restrict__ W3, const float* __restrict__ B3,
    const int*  __restrict__ resolutions,
    float* __restrict__ out, int N)
{
    // ---- LDS: weight fragments (f16, WMMA A layout) + biases + resolutions ----
    // frags 0..3  : W1^T  (M tile = f,            K = 0..31)
    // frags 4..11 : W2^T  (q = f-4: mtile = q>>1, kchunk = q&1)
    // frags 12..13: W3^T  (kchunk = f-12)
    __shared__ alignas(32) _Float16 lds_w[14 * 512];
    __shared__ float lds_b[160];     // [0,64) b1, [64,128) b2, [128,144) b3, [144,160) res(f32)

    const int tid = threadIdx.x;
#pragma unroll 1
    for (int idx = tid; idx < 14 * 512; idx += 128) {
        int f    = idx >> 9;
        int lane = (idx >> 4) & 31;
        int e    = idx & 15;
        int mloc = lane & 15;
        int kloc = e + ((lane >= 16) ? 16 : 0);  // A-layout K
        float v;
        if (f < 4) {
            v = W1[kloc * WIDTH + f * 16 + mloc];
        } else if (f < 12) {
            int q = f - 4, kc = q & 1, mt = q >> 1;
            v = W2[(kc * 32 + kloc) * WIDTH + mt * 16 + mloc];
        } else {
            int kc = f - 12;
            v = W3[(kc * 32 + kloc) * OUTD + mloc];
        }
        lds_w[idx] = (_Float16)v;
    }
    for (int i = tid; i < 160; i += 128) {
        float v;
        if      (i < 64)  v = B1[i];
        else if (i < 128) v = B2[i - 64];
        else if (i < 144) v = B3[i - 128];
        else              v = (float)resolutions[i - 144];
        lds_b[i] = v;
    }
    __syncthreads();

    const int lane = tid & 31;
    const int wv   = tid >> 5;
    const int hi   = lane >> 4;       // 0: lanes 0-15, 1: lanes 16-31
    const int m    = lane & 15;       // point within wave tile

    long pidx = (long)blockIdx.x * 64 + wv * 16 + m;
    long pc   = (pidx < N) ? pidx : (long)(N - 1);   // clamp, no divergence

    // ---- normalize coords ----
    float bbx = bb[0], bby = bb[1], bbz = bb[2];
    float px = (xyz[3 * pc + 0] - bbx) / (bb[3] - bbx);
    float py = (xyz[3 * pc + 1] - bby) / (bb[4] - bby);
    float pz = (xyz[3 * pc + 2] - bbz) / (bb[5] - bbz);

    // ---- hash-grid encode directly into the layer-1 B fragment ----
    // lane lo: levels 0..7 (features K=0..15); lane hi: levels 8..15 (K=16..31)
    const float2* __restrict__ tab2 = (const float2*)table;
    v16h bfeat;
    const int lbase = hi * 8;
#pragma unroll
    for (int j = 0; j < 8; ++j) {
        const int l = lbase + j;
        const float res = lds_b[144 + l];
        float sx = px * res, sy = py * res, sz = pz * res;
        float gx = floorf(sx), gy = floorf(sy), gz = floorf(sz);
        float rx = sx - gx, ry = sy - gy, rz = sz - gz;
        unsigned ix = (unsigned)(int)gx, iy = (unsigned)(int)gy, iz = (unsigned)(int)gz;
        float wx[2] = {1.f - rx, rx}, wy[2] = {1.f - ry, ry}, wz[2] = {1.f - rz, rz};
        const float2* __restrict__ tl = tab2 + (size_t)l * TSIZE;
        float f0 = 0.f, f1 = 0.f;
#pragma unroll
        for (int c = 0; c < 8; ++c) {
            unsigned ox = (c >> 2) & 1, oy = (c >> 1) & 1, oz = c & 1;
            unsigned h = ((ix + ox) * 1u) ^ ((iy + oy) * 2654435761u) ^ ((iz + oz) * 805459861u);
            h &= (TSIZE - 1u);
            float2 t = tl[h];                         // L2-resident gather (64 MB table)
            float w = wx[ox] * wy[oy] * wz[oz];
            f0 = fmaf(w, t.x, f0);
            f1 = fmaf(w, t.y, f1);
        }
        bfeat[2 * j]     = (_Float16)f0;
        bfeat[2 * j + 1] = (_Float16)f1;
    }

    // ---- layer 1: D[nt] = W1^T(tile nt) x X + b1, ReLU ----  (4 WMMAs, K=32)
    float d1[4][8];
#pragma unroll
    for (int nt = 0; nt < 4; ++nt) {
        v16h a = *(const v16h*)(lds_w + nt * 512 + lane * 16);
        v8f c;
#pragma unroll
        for (int r = 0; r < 8; ++r) c[r] = lds_b[nt * 16 + hi * 8 + r];
        c = WMMA_F16(a, bfeat, c);
#pragma unroll
        for (int r = 0; r < 8; ++r) d1[nt][r] = fmaxf(c[r], 0.f);
    }

    // ---- layer 2: 4 N-tiles x 2 K-chunks ----  (8 WMMAs)
    v16h bh[2];
    bh[0] = make_bfrag(d1[0], d1[1], hi);
    bh[1] = make_bfrag(d1[2], d1[3], hi);
    float d2[4][8];
#pragma unroll
    for (int mt = 0; mt < 4; ++mt) {
        v16h a0 = *(const v16h*)(lds_w + (4 + mt * 2 + 0) * 512 + lane * 16);
        v16h a1 = *(const v16h*)(lds_w + (4 + mt * 2 + 1) * 512 + lane * 16);
        v8f c;
#pragma unroll
        for (int r = 0; r < 8; ++r) c[r] = lds_b[64 + mt * 16 + hi * 8 + r];
        c = WMMA_F16(a0, bh[0], c);
        c = WMMA_F16(a1, bh[1], c);
#pragma unroll
        for (int r = 0; r < 8; ++r) d2[mt][r] = fmaxf(c[r], 0.f);
    }

    // ---- layer 3: 16 outputs, 2 K-chunks ----  (2 WMMAs)
    v16h bf3a = make_bfrag(d2[0], d2[1], hi);
    v16h bf3b = make_bfrag(d2[2], d2[3], hi);
    {
        v16h a0 = *(const v16h*)(lds_w + 12 * 512 + lane * 16);
        v16h a1 = *(const v16h*)(lds_w + 13 * 512 + lane * 16);
        v8f c;
#pragma unroll
        for (int r = 0; r < 8; ++r) c[r] = lds_b[128 + hi * 8 + r];
        c = WMMA_F16(a0, bf3a, c);
        c = WMMA_F16(a1, bf3b, c);

        // density = softplus(z[:,0]); channel 0 lives in vgpr 0 of lanes 0..15
        float z0 = c[0];
        float dens = (z0 > 0.f) ? (z0 + log1pf(__expf(-z0))) : log1pf(__expf(z0));
        if (hi == 0 && pidx < (long)N) out[pidx] = dens;
    }
}

extern "C" void kernel_launch(void* const* d_in, const int* in_sizes, int n_in,
                              void* d_out, int out_size, void* d_ws, size_t ws_size,
                              hipStream_t stream) {
    const float* xyz   = (const float*)d_in[0];
    const float* bb    = (const float*)d_in[1];
    const float* table = (const float*)d_in[2];
    const float* W1    = (const float*)d_in[3];
    const float* b1    = (const float*)d_in[4];
    const float* W2    = (const float*)d_in[5];
    const float* b2    = (const float*)d_in[6];
    const float* W3    = (const float*)d_in[7];
    const float* b3    = (const float*)d_in[8];
    const int*   res   = (const int*)d_in[9];
    float* out = (float*)d_out;

    int N = in_sizes[0] / 3;
    int blocks = (N + 63) / 64;          // 64 points per block (4 waves x 16)
    ngp_fused_kernel<<<blocks, 128, 0, stream>>>(
        xyz, bb, table, W1, b1, W2, b2, W3, b3, res, out, N);
}
